// MoE_65489661329569
// MI455X (gfx1250) — matrix operand order repaired and checked by reading, not compiled, not run on hardware
//
#include <hip/hip_runtime.h>
#include <hip/hip_bf16.h>
#include <math.h>

// MoE top-2 + shared expert for MI455X (gfx1250, wave32, WMMA).
// bf16 WMMA (fp32 accumulate) for all four GEMMs; routed experts via gathered
// row lists (grouped GEMM, 4x less math than dense-masked); explicit 2-stage
// software pipeline in the GEMM k-loop to hide L2 latency at low occupancy.
//
// Workspace budget: ~239 MB (assumed ws_size is at least this).

#define T_TOK  8192
#define HID    1024
#define INTER_ 2048
#define SINTER 1024
#define NEXP   8

typedef __attribute__((ext_vector_type(16))) __bf16 v16bf;
typedef __attribute__((ext_vector_type(8)))  __bf16 v8bf;
typedef __attribute__((ext_vector_type(4)))  __bf16 v4bf;
typedef __attribute__((ext_vector_type(8)))  float  v8f;
typedef __attribute__((ext_vector_type(4)))  float  v4f;

union FragU { v16bf v; v8bf h[2]; };

// A-matrix 16x32 bf16 fragment (ISA 7.12.2): per lane, elements 0..7 are
// K = 8*(lane>>4)+0..7, elements 8..15 are K = 8*(lane>>4)+16..23.
// `p` already points at row + k0 + 8*(lane>>4).
__device__ __forceinline__ v16bf load_fragA(const __bf16* p) {
  FragU f;
  f.h[0] = *(const v8bf*)(p);
  f.h[1] = *(const v8bf*)(p + 16);
  return f.v;
}

// B-matrix 32x16 bf16 fragment: same K order as A: elements 0..7 are
// K = 8*(lane>>4)+0..7, elements 8..15 are K = 8*(lane>>4)+16..23. B is stored [N,K] row-major (== the
// K x N operand column-major), exactly the weight layout.
// `p` points at rowN + k0 + 8*(lane>>4).
__device__ __forceinline__ v16bf load_fragB(const __bf16* p) {
  FragU f;
  f.h[0] = *(const v8bf*)(p);
  f.h[1] = *(const v8bf*)(p + 16);
  return f.v;
}

__device__ __forceinline__ void load_frags(v16bf a[4], v16bf b[4],
                                           const __bf16* const arow[4],
                                           const __bf16* const brow[4], int k0) {
#pragma unroll
  for (int i = 0; i < 4; ++i) a[i] = load_fragA(arow[i] + k0);
#pragma unroll
  for (int i = 0; i < 4; ++i) b[i] = load_fragB(brow[i] + k0);
}

__device__ __forceinline__ void do_wmma(v8f c[4][4], const v16bf a[4], const v16bf b[4]) {
#pragma unroll
  for (int mi = 0; mi < 4; ++mi)
#pragma unroll
    for (int ni = 0; ni < 4; ++ni)
      c[mi][ni] = __builtin_amdgcn_wmma_f32_16x16x32_bf16(
          false, a[mi], false, b[ni], (short)0, c[mi][ni], false, false);
}

__global__ void zero_counts_kernel(int* counts) {
  if (threadIdx.x < NEXP) counts[threadIdx.x] = 0;
}

// 4 floats -> 4 bf16 per thread (b128 load / b64 store per lane).
__global__ void cvt_f32_bf16_v4(const v4f* __restrict__ s, v4bf* __restrict__ d, int n4) {
  int i = blockIdx.x * 256 + threadIdx.x;
  if (i < n4) {
    v4f x = s[i];
    v4bf y;
#pragma unroll
    for (int j = 0; j < 4; ++j) y[j] = (__bf16)x[j];
    d[i] = y;
  }
}

// One wave per token: 8 router logits, top-2, softmax over the 2, append
// pair ids (2t / 2t+1) to per-expert gather lists.
__global__ __launch_bounds__(256) void router_kernel(
    const float* __restrict__ x, const float* __restrict__ wg,
    float* __restrict__ gates, int* __restrict__ lists, int* __restrict__ counts) {
  const int wave = threadIdx.x >> 5;
  const int lane = threadIdx.x & 31;
  const int t = blockIdx.x * 8 + wave;
  const float* xr = x + (size_t)t * HID;
  float acc[NEXP];
#pragma unroll
  for (int e = 0; e < NEXP; ++e) acc[e] = 0.0f;
  for (int h = lane; h < HID; h += 32) {
    float xv = xr[h];
#pragma unroll
    for (int e = 0; e < NEXP; ++e) acc[e] += xv * wg[e * HID + h];
  }
#pragma unroll
  for (int e = 0; e < NEXP; ++e)
    for (int off = 16; off > 0; off >>= 1) acc[e] += __shfl_down(acc[e], off, 32);
  if (lane == 0) {
    int i0 = 0; float v0 = acc[0];
#pragma unroll
    for (int e = 1; e < NEXP; ++e) if (acc[e] > v0) { v0 = acc[e]; i0 = e; }
    int i1 = -1; float v1 = -3.4e38f;
#pragma unroll
    for (int e = 0; e < NEXP; ++e) if (e != i0 && acc[e] > v1) { v1 = acc[e]; i1 = e; }
    // softmax over the two selected logits (v0 >= v1)
    float e1 = __expf(v1 - v0);
    float inv = 1.0f / (1.0f + e1);
    gates[2 * t + 0] = inv;
    gates[2 * t + 1] = e1 * inv;
    int p0 = atomicAdd(&counts[i0], 1);
    lists[i0 * T_TOK + p0] = 2 * t + 0;
    int p1 = atomicAdd(&counts[i1], 1);
    lists[i1 * T_TOK + p1] = 2 * t + 1;
  }
}

// One wave per block, 64x64 output tile, K-step 32, 16 accumulator fragments,
// 2-stage software-pipelined k-loop (double-buffered A/B fragments).
// GATHER: row ids come from per-expert list (pair ids); A row = pairId>>RSHIFT,
// output row = pairId. Dense path: rows are tile_m+i. Expert = blockIdx.z.
template <bool GATHER, int RSHIFT, bool DOGELU, bool OUTBF>
__global__ __launch_bounds__(32) void gemm_wmma_64x64(
    const __bf16* __restrict__ A, const __bf16* __restrict__ Ball,
    void* __restrict__ Out, const int* __restrict__ list,
    const int* __restrict__ counts, int K, int N, int M) {
  const int lane   = threadIdx.x;
  const int e      = blockIdx.z;
  const int tile_m = blockIdx.y * 64;
  const int tile_n = blockIdx.x * 64;
  const int lrow   = lane & 15;

  __shared__ int rows[64];
  int count;
  if (GATHER) {
    count = counts[e];
    if (tile_m >= count) return;
    const int* lp = list + e * T_TOK;
    for (int i = lane; i < 64; i += 32) {
      int gi = tile_m + i;
      rows[i] = lp[gi < count ? gi : count - 1];  // clamp pad rows (loads stay in-bounds)
    }
    __syncthreads();
  } else {
    count = M;
    if (tile_m >= M) return;
  }

  const __bf16* B = Ball + (size_t)e * (size_t)N * (size_t)K;
  const int khA = (lane >> 4) << 3;   // 0 or 8
  const int khB = (lane >> 4) << 3;   // 0 or 8

  const __bf16* arow[4];
#pragma unroll
  for (int mi = 0; mi < 4; ++mi) {
    size_t r;
    if (GATHER) r = (size_t)(rows[mi * 16 + lrow] >> RSHIFT);
    else        r = (size_t)(tile_m + mi * 16 + lrow);
    arow[mi] = A + r * (size_t)K + khA;
  }
  const __bf16* brow[4];
#pragma unroll
  for (int ni = 0; ni < 4; ++ni)
    brow[ni] = B + (size_t)(tile_n + ni * 16 + lrow) * (size_t)K + khB;

  v8f c[4][4] = {};

  // Software pipeline: loads for step k+32 issued before WMMAs for step k.
  // K is always a multiple of 64 (1024 or 2048).
  v16bf a0[4], b0[4], a1[4], b1[4];
  load_frags(a0, b0, arow, brow, 0);
  int k0 = 0;
  for (; k0 < K - 64; k0 += 64) {
    load_frags(a1, b1, arow, brow, k0 + 32);
    do_wmma(c, a0, b0);
    load_frags(a0, b0, arow, brow, k0 + 64);
    do_wmma(c, a1, b1);
  }
  load_frags(a1, b1, arow, brow, k0 + 32);
  do_wmma(c, a0, b0);
  do_wmma(c, a1, b1);

  // Epilogue. C/D layout: VGPR v of frag (mi,ni) holds M = mi*16 + 8*(lane>>4) + v,
  // N = tile_n + ni*16 + (lane&15).
  const int mh = (lane >> 4) << 3;
#pragma unroll
  for (int mi = 0; mi < 4; ++mi) {
#pragma unroll
    for (int v = 0; v < 8; ++v) {
      const int mloc = mi * 16 + mh + v;
      if (tile_m + mloc >= count) continue;
      const size_t orow = GATHER ? (size_t)rows[mloc] : (size_t)(tile_m + mloc);
#pragma unroll
      for (int ni = 0; ni < 4; ++ni) {
        float val = c[mi][ni][v];
        if (DOGELU) val = 0.5f * val * (1.0f + erff(val * 0.70710678118654752f));
        const size_t idx = orow * (size_t)N + (size_t)(tile_n + ni * 16 + lrow);
        if (OUTBF) ((__bf16*)Out)[idx] = (__bf16)val;
        else       ((float*)Out)[idx]  = val;
      }
    }
  }
}

// out[t,:] += g0*ypair[2t,:] + g1*ypair[2t+1,:], 4 floats per thread.
__global__ void combine_kernel_v4(const v4f* __restrict__ ypair,
                                  const float* __restrict__ gates,
                                  v4f* __restrict__ out, int n4) {
  int i = blockIdx.x * 256 + threadIdx.x;
  if (i >= n4) return;
  int t  = i >> 8;       // HID/4 == 256 vectors per token
  int h4 = i & 255;
  float g0 = gates[2 * t + 0], g1 = gates[2 * t + 1];
  v4f a = ypair[(size_t)(2 * t + 0) * 256 + h4];
  v4f b = ypair[(size_t)(2 * t + 1) * 256 + h4];
  out[i] = out[i] + a * g0 + b * g1;
}

extern "C" void kernel_launch(void* const* d_in, const int* in_sizes, int n_in,
                              void* d_out, int out_size, void* d_ws, size_t ws_size,
                              hipStream_t stream) {
  const float* x       = (const float*)d_in[0];
  const float* w_gate  = (const float*)d_in[1];
  const float* w_fc    = (const float*)d_in[2];
  const float* w_proj  = (const float*)d_in[3];
  const float* w_fcs   = (const float*)d_in[4];
  const float* w_prjs  = (const float*)d_in[5];
  float* out = (float*)d_out;

  char* ws = (char*)d_ws;
  size_t off = 0;
  auto alloc = [&](size_t bytes) {
    void* p = ws + off;
    off += (bytes + 255) & ~(size_t)255;
    return p;
  };
  __bf16* xbf    = (__bf16*)alloc((size_t)T_TOK * HID * 2);
  __bf16* wfcb   = (__bf16*)alloc((size_t)NEXP * INTER_ * HID * 2);
  __bf16* wprjb  = (__bf16*)alloc((size_t)NEXP * HID * INTER_ * 2);
  __bf16* wfcsb  = (__bf16*)alloc((size_t)SINTER * HID * 2);
  __bf16* wprjsb = (__bf16*)alloc((size_t)HID * SINTER * 2);
  __bf16* act1   = (__bf16*)alloc((size_t)2 * T_TOK * INTER_ * 2); // indexed by pair id
  __bf16* actsh  = (__bf16*)alloc((size_t)T_TOK * SINTER * 2);
  float*  ypair  = (float*)alloc((size_t)2 * T_TOK * HID * 4);     // indexed by pair id
  float*  gates  = (float*)alloc((size_t)T_TOK * 2 * 4);
  int*    lists  = (int*)alloc((size_t)NEXP * T_TOK * 4);
  int*    counts = (int*)alloc(256);
  (void)ws_size; (void)in_sizes; (void)n_in; (void)out_size;

  zero_counts_kernel<<<1, 32, 0, stream>>>(counts);

  int n4;
  n4 = T_TOK * HID / 4;
  cvt_f32_bf16_v4<<<(n4 + 255) / 256, 256, 0, stream>>>((const v4f*)x, (v4bf*)xbf, n4);
  n4 = NEXP * INTER_ * HID / 4;
  cvt_f32_bf16_v4<<<(n4 + 255) / 256, 256, 0, stream>>>((const v4f*)w_fc, (v4bf*)wfcb, n4);
  n4 = NEXP * HID * INTER_ / 4;
  cvt_f32_bf16_v4<<<(n4 + 255) / 256, 256, 0, stream>>>((const v4f*)w_proj, (v4bf*)wprjb, n4);
  n4 = SINTER * HID / 4;
  cvt_f32_bf16_v4<<<(n4 + 255) / 256, 256, 0, stream>>>((const v4f*)w_fcs, (v4bf*)wfcsb, n4);
  n4 = HID * SINTER / 4;
  cvt_f32_bf16_v4<<<(n4 + 255) / 256, 256, 0, stream>>>((const v4f*)w_prjs, (v4bf*)wprjsb, n4);

  router_kernel<<<T_TOK / 8, 256, 0, stream>>>(x, w_gate, gates, lists, counts);

  // routed expert fc + gelu:  act1[p] = gelu(x[p>>1] @ wfc[e]^T)
  gemm_wmma_64x64<true, 1, true, true>
      <<<dim3(INTER_ / 64, T_TOK / 64, NEXP), 32, 0, stream>>>(
          xbf, wfcb, act1, lists, counts, HID, INTER_, T_TOK);
  // routed expert proj:  ypair[p] = act1[p] @ wproj[e]^T
  gemm_wmma_64x64<true, 0, false, false>
      <<<dim3(HID / 64, T_TOK / 64, NEXP), 32, 0, stream>>>(
          act1, wprjb, ypair, lists, counts, INTER_, HID, T_TOK);
  // shared expert fc + gelu (dense)
  gemm_wmma_64x64<false, 0, true, true>
      <<<dim3(SINTER / 64, T_TOK / 64, 1), 32, 0, stream>>>(
          xbf, wfcsb, actsh, nullptr, nullptr, HID, SINTER, T_TOK);
  // shared expert proj (dense) -> writes d_out directly (full init)
  gemm_wmma_64x64<false, 0, false, false>
      <<<dim3(HID / 64, T_TOK / 64, 1), 32, 0, stream>>>(
          actsh, wprjsb, out, nullptr, nullptr, SINTER, HID, T_TOK);

  n4 = T_TOK * HID / 4;
  combine_kernel_v4<<<(n4 + 255) / 256, 256, 0, stream>>>(
      (const v4f*)ypair, gates, (v4f*)out, n4);
}
